// GAT_19774029431579
// MI455X (gfx1250) — matrix operand order repaired and checked by reading, not compiled, not run on hardware
//
#include <hip/hip_runtime.h>
#include <hip/hip_bf16.h>
#include <math.h>

// ---------------- problem constants (from reference) ----------------
#define F_IN   128
#define HID    64
#define HEADS  4
#define HC     256      // HEADS*HID
#define OUTC   16
#define NEG_SLOPE 0.2f

// GEMM tiling
#define BM 64           // M rows per block
#define KC 64           // K slab per LDS stage

// ---------------- WMMA types ----------------
typedef __attribute__((ext_vector_type(16))) __bf16 v16bf;
typedef __attribute__((ext_vector_type(8)))  float  v8f;
typedef __attribute__((ext_vector_type(4)))  unsigned int u32x4v;

union ABfrag { v16bf v; unsigned short u[16]; unsigned int w[8]; u32x4v q[2]; };

// fp32 -> bf16 (round to nearest even)
static __device__ __forceinline__ unsigned short f2bf(float f) {
    unsigned int u = __float_as_uint(f);
    unsigned int r = u + 0x7FFFu + ((u >> 16) & 1u);
    return (unsigned short)(r >> 16);
}

// float atomic max via monotonic integer ordering
static __device__ __forceinline__ void atomicMaxFloat(float* addr, float val) {
    if (val >= 0.0f) atomicMax((int*)addr, __float_as_int(val));
    else             atomicMin((unsigned int*)addr, __float_as_uint(val));
}

// ---------------- TDM 2D tile load: global -> LDS ----------------
#if __has_builtin(__builtin_amdgcn_tensor_load_to_lds)
#define USE_TDM 1
typedef unsigned int u32x4 __attribute__((ext_vector_type(4)));
typedef int          i32x8 __attribute__((ext_vector_type(8)));
typedef int          i32x4 __attribute__((ext_vector_type(4)));

// Load a tile_h x tile_w (elements, 2B each) tile whose rows are
// row_stride elements apart, from `gaddr` (tile start) into LDS at
// byte offset `lds_off` (packed tile_w-wide rows). OOB rows/cols
// (beyond tensor_w/tensor_h remaining extents) read as zero.
static __device__ __forceinline__ void tdm_load_2d_bf16(
        const void* gaddr, unsigned int lds_off,
        unsigned int tile_w, unsigned int tile_h,
        unsigned long long row_stride,
        unsigned long long tensor_w, unsigned long long tensor_h) {
    unsigned long long ga = (unsigned long long)gaddr;
    u32x4 g0;
    g0.x = 0x1u;                                         // count=1, user mode
    g0.y = lds_off;                                      // lds_addr (bytes)
    g0.z = (unsigned int)(ga & 0xFFFFFFFFull);           // global_addr[31:0]
    g0.w = (unsigned int)((ga >> 32) & 0x01FFFFFFull)    // global_addr[56:32]
         | 0x80000000u;                                  // type=2 ("image")
    i32x8 g1;
    g1[0] = 0x00010000;                                  // data_size=1 (2 bytes)
    g1[1] = (int)((tensor_w & 0xFFFFull) << 16);         // tensor_dim0[15:0]
    g1[2] = (int)(((tensor_w >> 16) & 0xFFFFull)         // tensor_dim0[31:16]
                | ((tensor_h & 0xFFFFull) << 16));       // tensor_dim1[15:0]
    g1[3] = (int)(((tensor_h >> 16) & 0xFFFFull)         // tensor_dim1[31:16]
                | ((unsigned long long)tile_w << 16));   // tile_dim0
    g1[4] = (int)(tile_h & 0xFFFFu);                     // tile_dim1 (tile_dim2=0)
    g1[5] = (int)(row_stride & 0xFFFFFFFFull);           // tensor_dim0_stride[31:0]
    g1[6] = (int)((row_stride >> 32) & 0xFFFFull);       // stride[47:32]
    g1[7] = 0;
    i32x4 z4 = {0, 0, 0, 0};
    i32x8 z8 = {0, 0, 0, 0, 0, 0, 0, 0};
    // 6-arg form (clang-23 / therock-10.0 headers): groups 2/3 NULL for 2D
    __builtin_amdgcn_tensor_load_to_lds(g0, g1, z4, z4, z8, 0);
}
#else
#define USE_TDM 0
#endif

// ---------------- utility kernels ----------------
__global__ void k_fill_f32(float* __restrict__ p, float v, long n) {
    long i = (long)blockIdx.x * blockDim.x + threadIdx.x;
    if (i < n) p[i] = v;
}

__global__ void k_cast_bf16(const float* __restrict__ src, unsigned short* __restrict__ dst, long n) {
    long i = (long)blockIdx.x * blockDim.x + threadIdx.x;
    if (i < n) dst[i] = f2bf(src[i]);
}

// weight cast + transpose: W [K x 256] f32 -> WT [256 x K] bf16
__global__ void k_castT_bf16(const float* __restrict__ W, unsigned short* __restrict__ WT, int K) {
    int i = blockIdx.x * blockDim.x + threadIdx.x;
    if (i >= HC * K) return;
    int n = i / K, k = i - n * K;
    WT[i] = f2bf(W[(size_t)k * HC + n]);
}

// h1 = acc + b1 (bias over 256 feats), cast to bf16
__global__ void k_bias_cast_bf16(const float* __restrict__ acc, const float* __restrict__ bias,
                                 unsigned short* __restrict__ dst, long n) {
    long i = (long)blockIdx.x * blockDim.x + threadIdx.x;
    if (i < n) dst[i] = f2bf(acc[i] + bias[i & (HC - 1)]);
}

// ---------------- bf16 WMMA GEMM: C[M x 256] = A[M x K] * W[K x 256] ----------------
// Weights are passed PRE-TRANSPOSED: WT [256 x K] bf16, so every WMMA B
// fragment is two contiguous 16B LDS reads (ds_load_b128).
// Block: 256 threads = 8 waves. Block macro-tile: 64 M rows x all 256 N cols.
// Wave w: m-tile = w>>1 (16 rows), n-half = w&1 (128 cols = 8 C tiles).
// A (BM x KC) and WT (256 x KC) slabs staged in LDS via TDM; one A fragment
// feeds 8 WMMAs. Fragment layouts per CDNA5 ISA 7.12.2 (wave32).
__global__ void __launch_bounds__(256)
k_gemm_bf16_wmma(const unsigned short* __restrict__ A,   // [M x K] bf16 bits
                 const unsigned short* __restrict__ WT,  // [256 x K] bf16 bits
                 float* __restrict__ C,                   // [M x 256] f32
                 int M, int K) {
    __shared__ __align__(16) unsigned short As [BM * KC];   // 8 KB, row-major rows of KC
    __shared__ __align__(16) unsigned short BsT[HC * KC];   // 32 KB, col-major: [ncol][KC]

    const int tid  = threadIdx.x;
    const int wave = tid >> 5;
    const int lane = tid & 31;
    const int row0 = blockIdx.x * BM;

    const int mt = wave >> 1;           // 0..3 : 16-row tile within block
    const int n0 = (wave & 1) * 128;    // 0 or 128

    v8f acc[8] = {};

    const int mA  = lane & 15;
    const int kbA = (lane < 16) ? 0 : 8;    // A: K 0-7 / 8-15 split per lane half
    const int nB  = lane & 15;
    const int kbB = (lane < 16) ? 0 : 8;    // B: K 0-7,16-23 / 8-15,24-31 split per lane half

#if USE_TDM
    const unsigned int ldsA = (unsigned int)(unsigned long long)(void*)As;
    const unsigned int ldsB = (unsigned int)(unsigned long long)(void*)BsT;
#endif

    for (int kc = 0; kc < K; kc += KC) {
#if USE_TDM
        if (tid == 0) {
            // A slab: BM x KC tile, rows K apart; zero-fill OOB rows of last block
            tdm_load_2d_bf16(A + (size_t)row0 * K + kc, ldsA,
                             KC, BM, (unsigned long long)K,
                             (unsigned long long)(K - kc),
                             (unsigned long long)(M - row0));
            // WT slab: 256 x KC tile (per-column K runs), rows K apart
            tdm_load_2d_bf16(WT + kc, ldsB,
                             KC, HC, (unsigned long long)K,
                             (unsigned long long)(K - kc),
                             (unsigned long long)HC);
        }
        __builtin_amdgcn_s_wait_tensorcnt(0);
#else
        for (int i = tid; i < BM * KC; i += 256) {
            int r = i / KC, c = i - r * KC;
            int row = row0 + r;
            As[i] = (row < M) ? A[(size_t)row * K + kc + c] : (unsigned short)0;
        }
        for (int i = tid; i < HC * KC; i += 256) {
            int ncol = i / KC, k = i - ncol * KC;
            BsT[i] = WT[(size_t)ncol * K + kc + k];
        }
#endif
        __syncthreads();

#pragma unroll
        for (int ks = 0; ks < KC; ks += 32) {
            // A fragment: two contiguous 16B runs (k = kbA..+7, 16+kbA..+7)
            ABfrag a;
            const unsigned short* ap = As + (mt * 16 + mA) * KC + ks;
            a.q[0] = *(const u32x4v*)(ap + kbA);
            a.q[1] = *(const u32x4v*)(ap + kbA + 16);
#pragma unroll
            for (int j = 0; j < 8; ++j) {
                // B fragment: two contiguous 16B runs (k = kbB..+7, 16+kbB..+7) of column
                ABfrag b;
                const unsigned short* bp = BsT + (n0 + j * 16 + nB) * KC + ks + kbB;
                b.q[0] = *(const u32x4v*)(bp);
                b.q[1] = *(const u32x4v*)(bp + 16);
                acc[j] = __builtin_amdgcn_wmma_f32_16x16x32_bf16(
                    false, a.v, false, b.v, (short)0, acc[j], false, false);
            }
        }
        __syncthreads();   // protect LDS before next slab DMA
    }

    // C/D layout: VGPR r -> M = r (lanes 0-15) / r+8 (lanes 16-31), N = lane&15
    const int nC = lane & 15;
    const int mh = (lane < 16) ? 0 : 8;
#pragma unroll
    for (int j = 0; j < 8; ++j) {
#pragma unroll
        for (int r = 0; r < 8; ++r) {
            int row = row0 + mt * 16 + r + mh;
            if (row < M) C[(size_t)row * HC + n0 + j * 16 + nC] = acc[j][r];
        }
    }
}

// ---------------- edge pass 1: score + leaky-relu + att dot + segment-max ----------------
static __device__ __forceinline__ void edge_sd(const int* __restrict__ ei, int e, int E, int N,
                                               int& s, int& d) {
    if (e < E) { s = ei[e]; d = ei[E + e]; }
    else       { s = d = e - E; }   // self loop
}

__global__ void k_edge_score(const float* __restrict__ xl, const float* __restrict__ xr,
                             const int* __restrict__ ei, const float* __restrict__ att,
                             float* __restrict__ sbuf, float* __restrict__ maxv,
                             int E, int N) {
    int idx = blockIdx.x * blockDim.x + threadIdx.x;
    int ET = E + N;
    if (idx >= ET * HEADS) return;
    int e = idx >> 2, h = idx & 3;
    int sn, dn; edge_sd(ei, e, E, N, sn, dn);
    const float* pl = xl + (size_t)sn * HC + h * HID;
    const float* pr = xr + (size_t)dn * HC + h * HID;
    const float* pa = att + h * HID;
    float acc = 0.0f;
#pragma unroll 8
    for (int c = 0; c < HID; ++c) {
        float v = pl[c] + pr[c];
        v = (v > 0.0f) ? v : NEG_SLOPE * v;
        acc += v * pa[c];
    }
    sbuf[idx] = acc;
    atomicMaxFloat(&maxv[dn * HEADS + h], acc);
}

// ---------------- edge pass 2: p = exp(s - max); segment-sum ----------------
__global__ void k_edge_exp(const int* __restrict__ ei, float* __restrict__ sbuf,
                           const float* __restrict__ maxv, float* __restrict__ denom,
                           int E, int N) {
    int idx = blockIdx.x * blockDim.x + threadIdx.x;
    int ET = E + N;
    if (idx >= ET * HEADS) return;
    int e = idx >> 2, h = idx & 3;
    int sn, dn; edge_sd(ei, e, E, N, sn, dn);
    float p = expf(sbuf[idx] - maxv[dn * HEADS + h]);
    sbuf[idx] = p;
    atomicAdd(&denom[dn * HEADS + h], p);
}

// ---------------- edge pass 3: weighted scatter-add aggregation ----------------
__global__ void k_edge_aggregate(const float* __restrict__ xl, const int* __restrict__ ei,
                                 const float* __restrict__ sbuf, const float* __restrict__ denom,
                                 float* __restrict__ acc, int E, int N) {
    int idx = blockIdx.x * blockDim.x + threadIdx.x;
    int ET = E + N;
    if (idx >= ET * HEADS) return;
    int e = idx >> 2, h = idx & 3;
    int sn, dn; edge_sd(ei, e, E, N, sn, dn);
    float alpha = sbuf[idx] / (denom[dn * HEADS + h] + 1e-16f);
    const float* pl = xl + (size_t)sn * HC + h * HID;
    float* po = acc + (size_t)dn * HC + h * HID;
#pragma unroll 4
    for (int c = 0; c < HID; ++c)
        atomicAdd(&po[c], alpha * pl[c]);
}

// ---------------- conv2 epilogue: mean over heads + bias ----------------
__global__ void k_head_mean(const float* __restrict__ acc, const float* __restrict__ b2,
                            float* __restrict__ h2, int N) {
    int idx = blockIdx.x * blockDim.x + threadIdx.x;
    if (idx >= N * HID) return;
    int n = idx / HID, c = idx - n * HID;
    const float* p = acc + (size_t)n * HC + c;
    float v = 0.25f * (p[0] + p[HID] + p[2 * HID] + p[3 * HID]);
    h2[idx] = v + b2[c];
}

// ---------------- global max pool per graph ----------------
__global__ void k_pool_max(const float* __restrict__ h2, const int* __restrict__ batch,
                           float* __restrict__ g, int N) {
    int idx = blockIdx.x * blockDim.x + threadIdx.x;
    if (idx >= N * HID) return;
    int n = idx / HID, c = idx - n * HID;
    atomicMaxFloat(&g[batch[n] * HID + c], h2[idx]);
}

// ---------------- classifier MLP: out = relu(g@Wc1+bc1)@Wc2+bc2 ----------------
__global__ void __launch_bounds__(64)
k_mlp(const float* __restrict__ g, const float* __restrict__ Wc1, const float* __restrict__ bc1,
      const float* __restrict__ Wc2, const float* __restrict__ bc2, float* __restrict__ out) {
    __shared__ float hsh[HID];
    int gid = blockIdx.x;     // graph index
    int t = threadIdx.x;      // 0..63
    float a = bc1[t];
    for (int k = 0; k < HID; ++k) {
        float gv = g[gid * HID + k];
        gv = isfinite(gv) ? gv : 0.0f;   // guard empty graphs
        a += gv * Wc1[k * HID + t];
    }
    hsh[t] = a > 0.0f ? a : 0.0f;
    __syncthreads();
    if (t < OUTC) {
        float o = bc2[t];
        for (int k = 0; k < HID; ++k) o += hsh[k] * Wc2[k * OUTC + t];
        out[gid * OUTC + t] = o;
    }
}

// ---------------- host-side launch ----------------
static inline int cdiv(long a, int b) { return (int)((a + b - 1) / b); }

extern "C" void kernel_launch(void* const* d_in, const int* in_sizes, int n_in,
                              void* d_out, int out_size, void* d_ws, size_t ws_size,
                              hipStream_t stream) {
    const float* x    = (const float*)d_in[0];
    const int*   ei   = (const int*)  d_in[1];
    const int*   bat  = (const int*)  d_in[2];
    const float* Wl1  = (const float*)d_in[3];
    const float* Wr1  = (const float*)d_in[4];
    const float* att1 = (const float*)d_in[5];
    const float* b1   = (const float*)d_in[6];
    const float* Wl2  = (const float*)d_in[7];
    const float* Wr2  = (const float*)d_in[8];
    const float* att2 = (const float*)d_in[9];
    const float* b2   = (const float*)d_in[10];
    const float* Wc1  = (const float*)d_in[11];
    const float* bc1  = (const float*)d_in[12];
    const float* Wc2  = (const float*)d_in[13];
    const float* bc2  = (const float*)d_in[14];

    const int N = in_sizes[0] / F_IN;      // 50000
    const int E = in_sizes[1] / 2;         // 600000
    const int G = out_size / OUTC;         // 64
    const int ET = E + N;

    // ---- workspace layout (bytes) ----
    char* ws = (char*)d_ws;
    size_t o = 0;
    unsigned short* xbf  = (unsigned short*)(ws + o); o += (size_t)N * F_IN * 2;  // reused as f32 h2
    unsigned short* wbf  = (unsigned short*)(ws + o); o += (size_t)HC * HC * 2;   // transposed weight scratch
    float* xl            = (float*)(ws + o);          o += (size_t)N * HC * 4;
    float* xr            = (float*)(ws + o);          o += (size_t)N * HC * 4;
    float* sbuf          = (float*)(ws + o);          o += (size_t)ET * HEADS * 4;
    float* maxv          = (float*)(ws + o);          o += (size_t)N * HEADS * 4;
    float* denom         = (float*)(ws + o);          o += (size_t)N * HEADS * 4;
    float* accb          = (float*)(ws + o);          o += (size_t)N * HC * 4;
    unsigned short* hbf  = (unsigned short*)(ws + o); o += (size_t)N * HC * 2;
    float* g             = (float*)(ws + o);          o += (size_t)G * HID * 4;
    float* h2 = (float*)xbf;   // reuse after GEMM1 consumed xbf (N*HID*4 == N*F_IN*2 bytes)

    const int TB = 256;
    const long nEdgeThreads = (long)ET * HEADS;

    // ---- conv1: bf16 cast + WMMA GEMMs (weights cast+transposed) ----
    k_cast_bf16<<<cdiv((long)N * F_IN, TB), TB, 0, stream>>>(x, xbf, (long)N * F_IN);
    k_castT_bf16<<<cdiv((long)HC * F_IN, TB), TB, 0, stream>>>(Wl1, wbf, F_IN);
    k_gemm_bf16_wmma<<<cdiv(N, BM), 256, 0, stream>>>(xbf, wbf, xl, N, F_IN);
    k_castT_bf16<<<cdiv((long)HC * F_IN, TB), TB, 0, stream>>>(Wr1, wbf, F_IN);
    k_gemm_bf16_wmma<<<cdiv(N, BM), 256, 0, stream>>>(xbf, wbf, xr, N, F_IN);

    // ---- conv1 edge passes ----
    k_fill_f32<<<cdiv((long)N * HEADS, TB), TB, 0, stream>>>(maxv, -INFINITY, (long)N * HEADS);
    (void)hipMemsetAsync(denom, 0, (size_t)N * HEADS * 4, stream);
    (void)hipMemsetAsync(accb, 0, (size_t)N * HC * 4, stream);
    k_edge_score<<<cdiv(nEdgeThreads, TB), TB, 0, stream>>>(xl, xr, ei, att1, sbuf, maxv, E, N);
    k_edge_exp<<<cdiv(nEdgeThreads, TB), TB, 0, stream>>>(ei, sbuf, maxv, denom, E, N);
    k_edge_aggregate<<<cdiv(nEdgeThreads, TB), TB, 0, stream>>>(xl, ei, sbuf, denom, accb, E, N);

    // h1 = acc + b1, cast to bf16 (input of conv2)
    k_bias_cast_bf16<<<cdiv((long)N * HC, TB), TB, 0, stream>>>(accb, b1, hbf, (long)N * HC);

    // ---- conv2: WMMA GEMMs (K=256), reuse xl/xr ----
    k_castT_bf16<<<cdiv((long)HC * HC, TB), TB, 0, stream>>>(Wl2, wbf, HC);
    k_gemm_bf16_wmma<<<cdiv(N, BM), 256, 0, stream>>>(hbf, wbf, xl, N, HC);
    k_castT_bf16<<<cdiv((long)HC * HC, TB), TB, 0, stream>>>(Wr2, wbf, HC);
    k_gemm_bf16_wmma<<<cdiv(N, BM), 256, 0, stream>>>(hbf, wbf, xr, N, HC);

    // ---- conv2 edge passes ----
    k_fill_f32<<<cdiv((long)N * HEADS, TB), TB, 0, stream>>>(maxv, -INFINITY, (long)N * HEADS);
    (void)hipMemsetAsync(denom, 0, (size_t)N * HEADS * 4, stream);
    (void)hipMemsetAsync(accb, 0, (size_t)N * HC * 4, stream);
    k_edge_score<<<cdiv(nEdgeThreads, TB), TB, 0, stream>>>(xl, xr, ei, att2, sbuf, maxv, E, N);
    k_edge_exp<<<cdiv(nEdgeThreads, TB), TB, 0, stream>>>(ei, sbuf, maxv, denom, E, N);
    k_edge_aggregate<<<cdiv(nEdgeThreads, TB), TB, 0, stream>>>(xl, ei, sbuf, denom, accb, E, N);

    // mean over heads + b2
    k_head_mean<<<cdiv((long)N * HID, TB), TB, 0, stream>>>(accb, b2, h2, N);

    // ---- global max pool + classifier ----
    k_fill_f32<<<cdiv((long)G * HID, TB), TB, 0, stream>>>(g, -INFINITY, (long)G * HID);
    k_pool_max<<<cdiv((long)N * HID, TB), TB, 0, stream>>>(h2, bat, g, N);
    k_mlp<<<G, 64, 0, stream>>>(g, Wc1, bc1, Wc2, bc2, (float*)d_out);
}
